// STGCN_50680614093554
// MI455X (gfx1250) — matrix operand order repaired and checked by reading, not compiled, not run on hardware
//
#include <hip/hip_runtime.h>
#include <float.h>

// ---------------------------------------------------------------- types
typedef __bf16 bf16_t;
typedef bf16_t v16bf __attribute__((ext_vector_type(16)));
typedef float  v8f   __attribute__((ext_vector_type(8)));

union Frag16 { v16bf v; unsigned int u[8]; };

static __device__ __forceinline__ bf16_t f2bf(float f) {
  unsigned u = __float_as_uint(f);
  unsigned r = (u + 0x7FFFu + ((u >> 16) & 1u)) >> 16;   // RNE
  unsigned short us = (unsigned short)r;
  bf16_t b;
  __builtin_memcpy(&b, &us, 2);
  return b;
}

// order-preserving float->uint map for atomicMax
static __device__ __forceinline__ unsigned fmapu(float f) {
  unsigned u = __float_as_uint(f);
  return (u & 0x80000000u) ? ~u : (u | 0x80000000u);
}
static __device__ __forceinline__ float funmap(unsigned m) {
  return (m & 0x80000000u) ? __uint_as_float(m & 0x7FFFFFFFu)
                           : __uint_as_float(~m);
}

#define NPB 16   // nodes per block in the conv kernels

// ---------------------------------------------------------------- utility
__global__ void zero_kernel(float* p, int n) {
  int i = blockIdx.x * blockDim.x + threadIdx.x;
  if (i < n) p[i] = 0.0f;
}
__global__ void fill_kernel(float* p, float v, int n) {
  int i = blockIdx.x * blockDim.x + threadIdx.x;
  if (i < n) p[i] = v;
}

// ---------------------------------------------------------------- input 1x1 conv (CIN=16 -> C=32)
__global__ void conv_in_kernel(const float* __restrict__ X,
                               const float* __restrict__ Win,  // [32][16]
                               const float* __restrict__ bin,  // [32]
                               float* __restrict__ out, int nNodes) {
  __shared__ float w[512];
  __shared__ float xs[NPB][128];
  int tid = threadIdx.x;
  for (int j = tid; j < 512; j += 256) w[j] = Win[j];
  int base = blockIdx.x * NPB;
  for (int nn = 0; nn < NPB; ++nn) {
    int n = base + nn;
    if (tid < 128) xs[nn][tid] = (n < nNodes) ? X[(size_t)n * 128 + tid] : 0.0f;
  }
  __syncthreads();
  int co = tid >> 3, t = tid & 7;
  for (int nn = 0; nn < NPB; ++nn) {
    int n = base + nn;
    if (n >= nNodes) break;
    float acc = bin[co];
    #pragma unroll
    for (int ci = 0; ci < 16; ++ci) acc += w[co * 16 + ci] * xs[nn][ci * 8 + t];
    out[(size_t)n * 256 + tid] = acc;
  }
}

// ---------------------------------------------------------------- gated conv (+optional residual) + BN stats
// FRONT: input raw, also computes residual. !FRONT: input gets bn1 affine, no residual.
template <bool FRONT>
__global__ void gated_kernel(const float* __restrict__ xin,
                             const float* __restrict__ inScale,  // [256] (back only)
                             const float* __restrict__ inShift,  // [256]
                             const float* __restrict__ gW,       // [3][32][32][3]
                             const float* __restrict__ gb,       // [3][32]
                             const float* __restrict__ resW,     // [32][32] (front)
                             const float* __restrict__ resb,     // [32]
                             float* __restrict__ resOut,
                             float* __restrict__ hOut,
                             float* __restrict__ sumC, float* __restrict__ sqC,
                             int nNodes) {
  __shared__ float ws[9216];
  __shared__ float wr[1024];
  __shared__ float xs[NPB][256];
  __shared__ float red[256];
  int tid = threadIdx.x;
  for (int j = tid; j < 9216; j += 256) ws[j] = gW[j];
  if (FRONT) for (int j = tid; j < 1024; j += 256) wr[j] = resW[j];
  int base = blockIdx.x * NPB;
  for (int nn = 0; nn < NPB; ++nn) {
    int n = base + nn;
    float v = 0.0f;
    if (n < nNodes) {
      v = xin[(size_t)n * 256 + tid];
      if (!FRONT) v = v * inScale[tid] + inShift[tid];
    }
    xs[nn][tid] = v;
  }
  __syncthreads();
  int co = tid >> 3, t = tid & 7;
  float bP = gb[co], bQ = gb[32 + co], bR = gb[64 + co];
  float lsum = 0.0f, lsq = 0.0f;
  for (int nn = 0; nn < NPB; ++nn) {
    int n = base + nn;
    if (n >= nNodes) break;
    const float* xrow = xs[nn];
    if (FRONT) {
      float acc = resb[co];
      #pragma unroll
      for (int ci = 0; ci < 32; ++ci) acc += wr[co * 32 + ci] * xrow[ci * 8 + t];
      resOut[(size_t)n * 256 + tid] = acc;
    }
    float P = bP, Q = bQ, R = bR;
    #pragma unroll 4
    for (int ci = 0; ci < 32; ++ci) {
      const float* wp = &ws[(co * 32 + ci) * 3];
      const float* wq = wp + 3072;
      const float* w2 = wp + 6144;
      float x0 = (t >= 1) ? xrow[ci * 8 + t - 1] : 0.0f;
      float x1 = xrow[ci * 8 + t];
      float x2 = (t <= 6) ? xrow[ci * 8 + t + 1] : 0.0f;
      P += wp[0] * x0 + wp[1] * x1 + wp[2] * x2;
      Q += wq[0] * x0 + wq[1] * x1 + wq[2] * x2;
      R += w2[0] * x0 + w2[1] * x1 + w2[2] * x2;
    }
    float sig = 1.0f / (1.0f + __expf(-Q));
    float h = fmaxf(P * sig + R, 0.0f);
    hOut[(size_t)n * 256 + tid] = h;
    lsum += h;
    lsq += h * h;
  }
  red[tid] = lsum;
  __syncthreads();
  if (t == 0) {
    float s = 0.0f;
    #pragma unroll
    for (int k = 0; k < 8; ++k) s += red[co * 8 + k];
    atomicAdd(&sumC[co], s);
  }
  __syncthreads();
  red[tid] = lsq;
  __syncthreads();
  if (t == 0) {
    float s = 0.0f;
    #pragma unroll
    for (int k = 0; k < 8; ++k) s += red[co * 8 + k];
    atomicAdd(&sqC[co], s);
  }
}

// ---------------------------------------------------------------- BN finalize: scale/shift from sums
__global__ void bn_finalize(const float* sum, const float* sq,
                            const float* g, const float* b,
                            float* scale, float* shift, float invCnt, int Cn) {
  int i = blockIdx.x * blockDim.x + threadIdx.x;
  if (i < Cn) {
    float m = sum[i] * invCnt;
    float v = sq[i] * invCnt - m * m;
    float sc = g[i] * rsqrtf(v + 1e-5f);
    scale[i] = sc;
    shift[i] = b[i] - m * sc;
  }
}

// ---------------------------------------------------------------- apply bn0 + convert to bf16 (per-channel affine)
__global__ void cvt_h_kernel(const float* __restrict__ h,
                             const float* __restrict__ scale,
                             const float* __restrict__ shift,
                             bf16_t* __restrict__ hbf, int total) {
  int i = blockIdx.x * blockDim.x + threadIdx.x;
  if (i < total) {
    int c = (i & 255) >> 3;
    hbf[i] = f2bf(h[i] * scale[c] + shift[c]);
  }
}

// ---------------------------------------------------------------- W (256x256, [fin][fout]) -> bf16 transposed [fout][fin]
__global__ void cvt_w_kernel(const float* __restrict__ W, bf16_t* __restrict__ Wt) {
  int i = blockIdx.x * blockDim.x + threadIdx.x;  // 65536
  int fo = i >> 8, fi = i & 255;
  Wt[i] = f2bf(W[fi * 256 + fo]);
}

// ---------------------------------------------------------------- WMMA GEMM: C[N,256] = A[N,256] @ W[256,256]
// Block = 256 threads = 8 waves; wave tile 16x16; block tile 32 rows x 64 cols.
__global__ void gemm_wmma_kernel(const bf16_t* __restrict__ A,
                                 const bf16_t* __restrict__ Bt,  // [fout][fin]
                                 float* __restrict__ C, int nRows) {
  int wave = threadIdx.x >> 5;
  int lane = threadIdx.x & 31;
  int mtile = wave & 1;
  int ntile = wave >> 1;
  int mbase = blockIdx.x * 32 + mtile * 16;
  int nbase = blockIdx.y * 64 + ntile * 16;
  int hv = lane >> 4;       // half-wave selector
  int l16 = lane & 15;
  int arow = mbase + l16;
  if (arow >= nRows) arow = nRows - 1;     // clamp: row outputs are independent
  const unsigned int* Ap = (const unsigned int*)(A + (size_t)arow * 256);
  const unsigned int* Bp = (const unsigned int*)(Bt + (size_t)(nbase + l16) * 256);
  v8f c;
  #pragma unroll
  for (int i = 0; i < 8; ++i) c[i] = 0.0f;
  #pragma unroll
  for (int kb = 0; kb < 256; kb += 32) {
    Frag16 a, b;
    #pragma unroll
    for (int vg = 0; vg < 8; ++vg) {
      int k0 = kb + (vg >> 2) * 16 + hv * 8 + (vg & 3) * 2;  // ISA A-layout 16x32 bf16
      a.u[vg] = Ap[k0 >> 1];
    }
    int bk = (kb + hv * 8) >> 1;                             // ISA B-layout 32x16 bf16
    #pragma unroll
    for (int vg = 0; vg < 8; ++vg) b.u[vg] = Bp[bk + (vg >> 2) * 8 + (vg & 3)];
    c = __builtin_amdgcn_wmma_f32_16x16x32_bf16(false, a.v, false, b.v,
                                                (short)0, c, false, false);
  }
  #pragma unroll
  for (int i = 0; i < 8; ++i) {
    int m = mbase + hv * 8 + i;                              // ISA C/D layout
    if (m < nRows) C[(size_t)m * 256 + nbase + l16] = c[i];
  }
}

// ---------------------------------------------------------------- degree / dinv
__global__ void deg_count_kernel(const int* __restrict__ dst, float* deg, int E) {
  int i = blockIdx.x * blockDim.x + threadIdx.x;
  if (i < E) atomicAdd(&deg[dst[i]], 1.0f);
}
__global__ void dinv_kernel(const float* deg, float* dinv, int n) {
  int i = blockIdx.x * blockDim.x + threadIdx.x;
  if (i < n) dinv[i] = rsqrtf(deg[i]);   // deg >= 1 (self loop) always
}

// ---------------------------------------------------------------- GCN aggregation
__global__ void agg_init_kernel(float* __restrict__ out, const float* __restrict__ bias, int total) {
  int i = blockIdx.x * blockDim.x + threadIdx.x;
  if (i < total) out[i] = bias[i & 255];
}
__global__ void agg_kernel(const float4* __restrict__ lin4, const float* __restrict__ dinv,
                           const int* __restrict__ src, const int* __restrict__ dst,
                           float* __restrict__ out, int E, int N) {
  int i = blockIdx.x * blockDim.x + threadIdx.x;
  int total = (E + N) * 64;
  if (i >= total) return;
  int e = i >> 6, f4 = i & 63;
  int s, d;
  if (e < E) { s = src[e]; d = dst[e]; }
  else       { s = e - E;  d = s; }
  float w = dinv[s] * dinv[d];
  float4 v = lin4[(size_t)s * 64 + f4];
  float* o = out + (size_t)d * 256 + f4 * 4;
  atomicAdd(o + 0, v.x * w);
  atomicAdd(o + 1, v.y * w);
  atomicAdd(o + 2, v.z * w);
  atomicAdd(o + 3, v.w * w);
}

// ---------------------------------------------------------------- per-feature BN reduce (256 features over N)
__global__ void bn_reduce_feat(const float* __restrict__ h, float* sum, float* sq, int nNodes) {
  int f = threadIdx.x;
  float s = 0.0f, q = 0.0f;
  for (int n = blockIdx.x; n < nNodes; n += gridDim.x) {
    float v = h[(size_t)n * 256 + f];
    s += v;
    q += v * v;
  }
  atomicAdd(&sum[f], s);
  atomicAdd(&sq[f], q);
}

// ---------------------------------------------------------------- bn2 apply + residual (+relu)
__global__ void combine_kernel(const float* __restrict__ h,
                               const float* __restrict__ scale,
                               const float* __restrict__ shift,
                               const float* __restrict__ res,
                               float* __restrict__ xout, int doRelu, int total) {
  int i = blockIdx.x * blockDim.x + threadIdx.x;
  if (i < total) {
    int c = (i & 255) >> 3;
    float v = h[i] * scale[c] + shift[c] + res[i];
    if (doRelu) v = fmaxf(v, 0.0f);
    xout[i] = v;
  }
}

// ---------------------------------------------------------------- graph pooling
__global__ void pool_init_kernel(float* cnt, float* spool, unsigned* mpool) {
  int i = blockIdx.x * blockDim.x + threadIdx.x;
  if (i < 2048) { spool[i] = 0.0f; mpool[i] = fmapu(-FLT_MAX); }
  if (i < 64) cnt[i] = 0.0f;
}
__global__ void pool_kernel(const float* __restrict__ x, const int* __restrict__ graphs,
                            float* cnt, float* spool, unsigned* mpool, int nNodes) {
  int i = blockIdx.x * blockDim.x + threadIdx.x;
  int n = i >> 5, c = i & 31;
  if (n >= nNodes) return;
  int g = graphs[n];
  float v = x[(size_t)n * 256 + c * 8 + 7];  // t = T-1
  atomicAdd(&spool[g * 32 + c], v);
  atomicMax(&mpool[g * 32 + c], fmapu(v));
  if (c == 0) atomicAdd(&cnt[g], 1.0f);
}

// ---------------------------------------------------------------- readout MLP 96 -> 128 -> 64 -> 1
__global__ void mlp_kernel(const float* __restrict__ cnt, const float* __restrict__ spool,
                           const unsigned* __restrict__ mpool,
                           const float* __restrict__ W1, const float* __restrict__ b1,
                           const float* __restrict__ W2, const float* __restrict__ b2,
                           const float* __restrict__ W3, const float* __restrict__ b3,
                           float* __restrict__ out) {
  __shared__ float xg[96], h1[128], h2[64];
  int g = blockIdx.x, tid = threadIdx.x;
  if (tid < 32) {
    float c = cnt[g];
    float s = spool[g * 32 + tid];
    xg[tid] = s / fmaxf(c, 1.0f);
    float mv = funmap(mpool[g * 32 + tid]);
    xg[32 + tid] = (c > 0.0f) ? mv : 0.0f;
    xg[64 + tid] = s;
  }
  __syncthreads();
  float a = b1[tid];
  for (int k = 0; k < 96; ++k) a += xg[k] * W1[k * 128 + tid];
  h1[tid] = fmaxf(a, 0.0f);
  __syncthreads();
  if (tid < 64) {
    float a2 = b2[tid];
    for (int k = 0; k < 128; ++k) a2 += h1[k] * W2[k * 64 + tid];
    h2[tid] = fmaxf(a2, 0.0f);
  }
  __syncthreads();
  if (tid == 0) {
    float a3 = b3[0];
    for (int k = 0; k < 64; ++k) a3 += h2[k] * W3[k];
    out[g] = a3;
  }
}

// ---------------------------------------------------------------- host orchestration
static inline int cdiv(int a, int b) { return (a + b - 1) / b; }

extern "C" void kernel_launch(void* const* d_in, const int* in_sizes, int n_in,
                              void* d_out, int out_size, void* d_ws, size_t ws_size,
                              hipStream_t stream) {
  const float* X      = (const float*)d_in[0];
  const int*   edge   = (const int*)d_in[1];
  const int*   graphs = (const int*)d_in[2];
  const float* W_in   = (const float*)d_in[3];
  const float* b_in   = (const float*)d_in[4];
  const float* res_W  = (const float*)d_in[5];
  const float* res_b  = (const float*)d_in[6];
  const float* g1_W   = (const float*)d_in[7];
  const float* g1_b   = (const float*)d_in[8];
  const float* bn0_g  = (const float*)d_in[9];
  const float* bn0_b  = (const float*)d_in[10];
  const float* gcn_W  = (const float*)d_in[11];
  const float* gcn_b  = (const float*)d_in[12];
  const float* bn1_g  = (const float*)d_in[13];
  const float* bn1_b  = (const float*)d_in[14];
  const float* g2_W   = (const float*)d_in[15];
  const float* g2_b   = (const float*)d_in[16];
  const float* bn2_g  = (const float*)d_in[17];
  const float* bn2_b  = (const float*)d_in[18];
  const float* oW1    = (const float*)d_in[19];
  const float* ob1    = (const float*)d_in[20];
  const float* oW2    = (const float*)d_in[21];
  const float* ob2    = (const float*)d_in[22];
  const float* oW3    = (const float*)d_in[23];
  const float* ob3    = (const float*)d_in[24];
  float* out = (float*)d_out;

  const int Nn = in_sizes[2];        // 50000
  const int En = in_sizes[1] / 2;    // 400000
  const int NF = Nn * 256;

  char* ws = (char*)d_ws;
  size_t off = 0;
  auto alloc = [&](size_t bytes) -> char* {
    char* p = ws + off;
    off = (off + bytes + 255) & ~(size_t)255;
    return p;
  };
  float*  bufX   = (float*)alloc((size_t)NF * 4);
  float*  bufRes = (float*)alloc((size_t)NF * 4);
  float*  bufH   = (float*)alloc((size_t)NF * 4);  // gated1 out -> GEMM out -> gated2 out
  float*  bufG   = (float*)alloc((size_t)NF * 4);  // aggregation out
  bf16_t* hbf    = (bf16_t*)alloc((size_t)NF * 2);
  bf16_t* wbf    = (bf16_t*)alloc(65536 * 2);
  float*  deg    = (float*)alloc((size_t)Nn * 4);
  float*  dinv   = (float*)alloc((size_t)Nn * 4);
  float*  stats  = (float*)alloc(8192 * 4);

  float* sum0 = stats;        float* sq0 = stats + 32;
  float* sc0  = stats + 64;   float* sh0 = stats + 96;
  float* sum1 = stats + 128;  float* sq1 = stats + 384;
  float* sc1  = stats + 640;  float* sh1 = stats + 896;
  float* sum2 = stats + 1152; float* sq2 = stats + 1184;
  float* sc2  = stats + 1216; float* sh2 = stats + 1248;
  float* cntP = stats + 1280;
  float* spool = stats + 1344;
  unsigned* mpool = (unsigned*)(stats + 3392);

  const int* srcA = edge;
  const int* dstA = edge + En;

  // degrees (layer-invariant): deg = 1 (self loop) + in-degree
  fill_kernel<<<cdiv(Nn, 256), 256, 0, stream>>>(deg, 1.0f, Nn);
  deg_count_kernel<<<cdiv(En, 256), 256, 0, stream>>>(dstA, deg, En);
  dinv_kernel<<<cdiv(Nn, 256), 256, 0, stream>>>(deg, dinv, Nn);

  // input conv
  conv_in_kernel<<<cdiv(Nn, NPB), 256, 0, stream>>>(X, W_in, b_in, bufX, Nn);

  for (int l = 0; l < 2; ++l) {
    zero_kernel<<<cdiv(1280, 256), 256, 0, stream>>>(stats, 1280);

    // residual + gated conv 1 + bn0 stats
    gated_kernel<true><<<cdiv(Nn, NPB), 256, 0, stream>>>(
        bufX, nullptr, nullptr, g1_W + l * 9216, g1_b + l * 96,
        res_W + l * 1024, res_b + l * 32, bufRes, bufH, sum0, sq0, Nn);
    bn_finalize<<<1, 32, 0, stream>>>(sum0, sq0, bn0_g + l * 32, bn0_b + l * 32,
                                      sc0, sh0, 1.0f / (float)(Nn * 8), 32);

    // bn0 apply + bf16 pack; pack W transposed
    cvt_h_kernel<<<cdiv(NF, 256), 256, 0, stream>>>(bufH, sc0, sh0, hbf, NF);
    cvt_w_kernel<<<256, 256, 0, stream>>>(gcn_W + l * 65536, wbf);

    // WMMA GEMM: lin = h @ W  (overwrites bufH)
    dim3 gg(cdiv(Nn, 32), 4);
    gemm_wmma_kernel<<<gg, 256, 0, stream>>>(hbf, wbf, bufH, Nn);

    // aggregation: bufG = bias; += norm-weighted neighbor features
    agg_init_kernel<<<cdiv(NF, 256), 256, 0, stream>>>(bufG, gcn_b + l * 256, NF);
    agg_kernel<<<cdiv((En + Nn) * 64, 256), 256, 0, stream>>>(
        (const float4*)bufH, dinv, srcA, dstA, bufG, En, Nn);

    // bn1 stats (per-feature over N)
    bn_reduce_feat<<<512, 256, 0, stream>>>(bufG, sum1, sq1, Nn);
    bn_finalize<<<1, 256, 0, stream>>>(sum1, sq1, bn1_g + l * 256, bn1_b + l * 256,
                                       sc1, sh1, 1.0f / (float)Nn, 256);

    // bn1 apply (fused) + gated conv 2 + bn2 stats  (overwrites bufH)
    gated_kernel<false><<<cdiv(Nn, NPB), 256, 0, stream>>>(
        bufG, sc1, sh1, g2_W + l * 9216, g2_b + l * 96,
        nullptr, nullptr, nullptr, bufH, sum2, sq2, Nn);
    bn_finalize<<<1, 32, 0, stream>>>(sum2, sq2, bn2_g + l * 32, bn2_b + l * 32,
                                      sc2, sh2, 1.0f / (float)(Nn * 8), 32);

    // bn2 apply + residual (+relu except last layer)
    combine_kernel<<<cdiv(NF, 256), 256, 0, stream>>>(bufH, sc2, sh2, bufRes, bufX,
                                                      (l != 1) ? 1 : 0, NF);
  }

  // readout
  pool_init_kernel<<<cdiv(2048, 256), 256, 0, stream>>>(cntP, spool, mpool);
  pool_kernel<<<cdiv(Nn * 32, 256), 256, 0, stream>>>(bufX, graphs, cntP, spool, mpool, Nn);
  mlp_kernel<<<64, 128, 0, stream>>>(cntP, spool, mpool, oW1, ob1, oW2, ob2, oW3, ob3, out);
}
